// ConvSplitTree_42949672961883
// MI455X (gfx1250) — matrix operand order repaired and checked by reading, not compiled, not run on hardware
//
#include <hip/hip_runtime.h>

// ---------------------------------------------------------------------------
// Fused decision-tree conv for MI455X (gfx1250, wave32, WMMA).
//
//   out[n,h,w] = (sum_l pred_l + sum_l b_l)/64 + pred_leaf + b_leaf
//   pred_l = conv3x3(data, w_pred[l]),  leaf from 6-ch split conv on x.
//
// Implicit GEMM with v_wmma_f32_16x16x32_bf16:
//   A = w_pred tile  (M=16 leaves x K=32 ch)   [bf16]
//   B = im2col data  (K=32 ch    x N=16 pixels)[bf16]
//   K-chunks = the 9 conv taps.
//
// Blocking: each wave owns a 64-leaf x 64-pixel macro-tile (16 f32 accum
// fragments, 128 VGPRs). Tap loop is OUTER and rolled: per tap load the 4 A
// fragments once, stream 4 B fragments, 16 WMMAs. A-fragment reuse is thus
// structural (no LICM hoisting -> no scratch spills as in the previous rev).
// One 256-thread workgroup per (n,h) row; 3 halo rows + all weights staged
// in LDS (~136 KB of 320 KB/WGP => 2 workgroups resident).
// ---------------------------------------------------------------------------

typedef __attribute__((ext_vector_type(16))) __bf16 v16bf;
typedef __attribute__((ext_vector_type(8)))  __bf16 v8bf;
typedef __attribute__((ext_vector_type(8)))  float  v8f;

#define NB    4
#define CIN   32
#define HH_   512
#define WW_   512
#define NLEAF 64
#define COLS  514                       // W + 2 halo columns
#define LDS_DATA_ELEMS (3 * COLS * CIN) // 49344 bf16
#define LDS_W_ELEMS    (9 * NLEAF * CIN)// 18432 bf16
#define SMEM_BYTES (LDS_DATA_ELEMS * 2 + LDS_W_ELEMS * 2 + 56 * 4)

__global__ __launch_bounds__(256, 4)
void tree_conv_fused(const float* __restrict__ x,
                     const float* __restrict__ data,
                     const float* __restrict__ w_split,
                     const float* __restrict__ w_pred,
                     const float* __restrict__ b_pred,
                     float* __restrict__ out)
{
    extern __shared__ __align__(64) char smem[];
    __bf16* lds_data = (__bf16*)smem;                  // [3][514][32]  (row,col,c)
    __bf16* lds_w    = lds_data + LDS_DATA_ELEMS;      // [9][64][32]   (tap,leaf,c)
    float*  lds_ws   = (float*)(lds_w + LDS_W_ELEMS);  // [6][9] normalized split w
    float*  lds_misc = lds_ws + 54;                    // [0] = sum(b_pred)

    const int tid = threadIdx.x;
    const int n   = blockIdx.x / HH_;
    const int h   = blockIdx.x % HH_;

    // ---- stage data rows h-1..h+1 (fp32 -> bf16), coalesced global reads ----
    for (int idx = tid; idx < LDS_DATA_ELEMS; idx += 256) {
        int c   = idx / (3 * COLS);
        int rem = idx - c * (3 * COLS);
        int r   = rem / COLS;
        int col = rem - r * COLS;
        int hh  = h + r - 1;
        int ww  = col - 1;
        float v = 0.0f;
        if (hh >= 0 && hh < HH_ && ww >= 0 && ww < WW_)
            v = data[(((size_t)n * CIN + c) * HH_ + hh) * WW_ + ww];
        lds_data[(r * COLS + col) * CIN + c] = (__bf16)v;
    }
    // ---- stage prediction weights, scatter to [tap][leaf][c] ----------------
    for (int g = tid; g < LDS_W_ELEMS; g += 256) {
        int leaf = g / (CIN * 9);
        int rem  = g - leaf * (CIN * 9);
        int c    = rem / 9;
        int t    = rem - c * 9;
        lds_w[(t * NLEAF + leaf) * CIN + c] = (__bf16)w_pred[g];
    }
    // ---- normalized split weights (relu; +0.1/9 if filter sum < 0.1) --------
    if (tid < 6) {
        float wf[9], s = 0.0f;
        for (int i = 0; i < 9; ++i) {
            float v = w_split[tid * 9 + i];
            v = v > 0.0f ? v : 0.0f;
            wf[i] = v; s += v;
        }
        float add = (s < 0.1f) ? (0.1f / 9.0f) : 0.0f;
        for (int i = 0; i < 9; ++i) lds_ws[tid * 9 + i] = wf[i] + add;
    }
    if (tid == 0) {
        float s = 0.0f;
        for (int l = 0; l < NLEAF; ++l) s += b_pred[l];
        lds_misc[0] = s;
    }
    __syncthreads();

    const int   lane  = tid & 31;
    const int   p     = lane & 15;          // pixel column within tile (N index)
    const int   khalf = lane >> 4;          // lane half selects K/M sub-block
    const int   wbase = (tid >> 5) * 64;    // wave's 64-pixel macro-tile
    const float bsum  = lds_misc[0];

    // 16 accumulator fragments: acc[pixel-tile][leaf-tile]
    v8f acc[4][4] = {};

    // ---- main WMMA stream: taps outer (rolled), A loaded once per tap ------
#pragma unroll 1
    for (int t = 0; t < 9; ++t) {
        const int row  = t / 3;          // dh + 1
        const int dcol = t - row * 3;    // dw + 1

        // A fragments: K = half*16 + e
        v16bf a[4];
#pragma unroll
        for (int m = 0; m < 4; ++m) {
            const __bf16* wrow = &lds_w[(t * NLEAF + m * 16 + p) * CIN + khalf * 16];
            v8bf lo = *(const v8bf*)(wrow);
            v8bf hi = *(const v8bf*)(wrow + 8);
            a[m] = __builtin_shufflevector(lo, hi,
                        0,1,2,3,4,5,6,7,8,9,10,11,12,13,14,15);
        }
        // stream B fragments for the 4 pixel tiles
#pragma unroll
        for (int i = 0; i < 4; ++i) {
            const v16bf b = *(const v16bf*)
                &lds_data[((row * COLS) + (wbase + i * 16 + p + dcol)) * CIN
                          + khalf * 16];
#pragma unroll
            for (int m = 0; m < 4; ++m)
                acc[i][m] = __builtin_amdgcn_wmma_f32_16x16x32_bf16(
                    false, a[m], false, b, (short)0, acc[i][m], false, false);
        }
    }

    // ---- per pixel tile: leaf index + reduction + store --------------------
#pragma unroll
    for (int i = 0; i < 4; ++i) {
        const int wpix = wbase + i * 16 + p;

        // split conv (sigmoid folded into threshold: v < ln(1.5))
        float vc0 = 0, vc1 = 0, vc2 = 0, vc3 = 0, vc4 = 0, vc5 = 0;
#pragma unroll
        for (int kh = 0; kh < 3; ++kh) {
            int hh = h + kh - 1;
            bool hok = (hh >= 0) && (hh < HH_);
#pragma unroll
            for (int kw = 0; kw < 3; ++kw) {
                int ww = wpix + kw - 1;
                float xv = 0.0f;
                if (hok && ww >= 0 && ww < WW_)
                    xv = x[((size_t)n * HH_ + hh) * WW_ + ww];
                int ki = kh * 3 + kw;
                vc0 += xv * lds_ws[0 * 9 + ki];
                vc1 += xv * lds_ws[1 * 9 + ki];
                vc2 += xv * lds_ws[2 * 9 + ki];
                vc3 += xv * lds_ws[3 * 9 + ki];
                vc4 += xv * lds_ws[4 * 9 + ki];
                vc5 += xv * lds_ws[5 * 9 + ki];
            }
        }
        const float TH = 0.4054651081081644f; // logit(0.6)
        int leaf = 0;
        leaf |= (vc0 < TH) ? 32 : 0;
        leaf |= (vc1 < TH) ? 16 : 0;
        leaf |= (vc2 < TH) ?  8 : 0;
        leaf |= (vc3 < TH) ?  4 : 0;
        leaf |= (vc4 < TH) ?  2 : 0;
        leaf |= (vc5 < TH) ?  1 : 0;

        // reduce: this lane holds M = mtile*16 + khalf*8 + j
        float tot = 0.0f, sel = 0.0f;
#pragma unroll
        for (int j = 0; j < 8; ++j) {
            int mrow = khalf * 8 + j;
            tot += acc[i][0][j] + acc[i][1][j] + acc[i][2][j] + acc[i][3][j];
            sel += (leaf == (mrow     )) ? acc[i][0][j] : 0.0f;
            sel += (leaf == (mrow + 16)) ? acc[i][1][j] : 0.0f;
            sel += (leaf == (mrow + 32)) ? acc[i][2][j] : 0.0f;
            sel += (leaf == (mrow + 48)) ? acc[i][3][j] : 0.0f;
        }
        tot += __shfl_xor(tot, 16, 32);   // combine the two lane halves (wave32)
        sel += __shfl_xor(sel, 16, 32);

        if (khalf == 0) {
            out[((size_t)n * HH_ + h) * WW_ + wpix] =
                (tot + bsum) * (1.0f / 64.0f) + sel + b_pred[leaf];
        }
    }
}

extern "C" void kernel_launch(void* const* d_in, const int* in_sizes, int n_in,
                              void* d_out, int out_size, void* d_ws, size_t ws_size,
                              hipStream_t stream)
{
    const float* x       = (const float*)d_in[0];
    const float* data    = (const float*)d_in[1];
    const float* w_split = (const float*)d_in[2];
    const float* w_pred  = (const float*)d_in[3];
    const float* b_pred  = (const float*)d_in[4];
    float*       out     = (float*)d_out;
    (void)in_sizes; (void)n_in; (void)out_size; (void)d_ws; (void)ws_size;

    dim3 grid(NB * HH_);   // one workgroup per (n, h) row
    dim3 block(256);       // 8 wave32 waves
    tree_conv_fused<<<grid, block, SMEM_BYTES, stream>>>(
        x, data, w_split, w_pred, b_pred, out);
}
